// NTMCell_43748536877126
// MI455X (gfx1250) — matrix operand order repaired and checked by reading, not compiled, not run on hardware
//
#include <hip/hip_runtime.h>
#include <hip/hip_bf16.h>

// NTM cell, MI455X (gfx1250, wave32). HBM-store bound on m_t (268 MB);
// GEMMs via v_wmma_f32_16x16x32_f16 (f16 in, f32 acc); m0 staged to LDS
// with GLOBAL_LOAD_ASYNC_TO_LDS_B128 overlapped against the gates GEMM.

#define BATCH 4096
#define UDIM  256
#define NSLOT 256
#define MDIM  64
#define HPAD  272   // 268 head cols padded to 17 tiles of 16

typedef __attribute__((ext_vector_type(16))) _Float16 v16h;
typedef __attribute__((ext_vector_type(8)))  _Float16 v8h;
typedef __attribute__((ext_vector_type(8)))  float    v8f;
typedef __attribute__((ext_vector_type(4)))  int      v4i;

#if defined(__has_builtin)
#if __has_builtin(__builtin_amdgcn_global_load_async_to_lds_b128) && \
    __has_builtin(__builtin_amdgcn_s_wait_asynccnt)
#define USE_ASYNC_LDS 1
#endif
#endif

typedef __attribute__((address_space(1))) v4i* gvp4;
typedef __attribute__((address_space(3))) v4i* lvp4;

union HF16 { v16h v; v8h h[2]; };

__device__ __forceinline__ float sigf(float x) { return 1.0f / (1.0f + __expf(-x)); }
__device__ __forceinline__ float softplusf(float x) { return (x > 20.0f) ? x : log1pf(__expf(x)); }

__device__ __forceinline__ float wredsum(float v) {
#pragma unroll
  for (int o = 16; o > 0; o >>= 1) v += __shfl_xor(v, o, 32);
  return v;
}
__device__ __forceinline__ float wredmax(float v) {
#pragma unroll
  for (int o = 16; o > 0; o >>= 1) v = fmaxf(v, __shfl_xor(v, o, 32));
  return v;
}

// A-fragment (16x32 f16) from LDS, row-major with rowStride halfs.
// ISA layout: lane L row=L&15; VGPR v (half pair) k = hi_half*16 + (L>>4)*8 + j
__device__ __forceinline__ v16h lds_a16(const _Float16* arr, int rowStride, int k0, int lane) {
  int row = lane & 15, hi = lane >> 4;
  const _Float16* p = arr + row * rowStride + k0 + hi * 8;
  HF16 u;
  u.h[0] = *(const v8h*)p;
  u.h[1] = *(const v8h*)(p + 16);
  return u.v;
}
// B-fragment (32x16 f16) from global weight stored N-major: W[n][K].
// lane L col=L&15, halfs hold k = (L>>4)*8 + 0..7, then 16 + (L>>4)*8 + 0..7.
__device__ __forceinline__ v16h glb_b16(const _Float16* W, int K, int n0, int k0, int lane) {
  int col = lane & 15, hi = lane >> 4;
  const _Float16* p = W + (size_t)(n0 + col) * K + (k0 + hi * 8);
  HF16 u;
  u.h[0] = *(const v8h*)p;
  u.h[1] = *(const v8h*)(p + 16);
  return u.v;
}
__device__ __forceinline__ v8f wmma16(v16h a, v16h b, v8f c) {
  return __builtin_amdgcn_wmma_f32_16x16x32_f16(false, a, false, b, (short)0, c, false, false);
}

// ---------------- prep: transpose + f16-convert weights ----------------
__global__ __launch_bounds__(256) void ntm_prep_w(
    const float* __restrict__ Wx, const float* __restrict__ W_r,
    const float* __restrict__ W_w, const float* __restrict__ W_ou,
    _Float16* __restrict__ wxt, _Float16* __restrict__ wht, _Float16* __restrict__ wot)
{
  int stride = gridDim.x * 256;
  int tid = blockIdx.x * 256 + threadIdx.x;
  // wxt[n*256+k] = Wx[k*1024+n], n<1024,k<256  (top half of Wx)
  for (int i = tid; i < 1024 * 256; i += stride) {
    int n = i >> 8, k = i & 255;
    wxt[i] = (_Float16)Wx[k * 1024 + n];
  }
  // wht[n*256+k]: n<70 -> W_r[k*70+n]; 70<=n<268 -> W_w[k*198+(n-70)]; else 0
  for (int i = tid; i < HPAD * 256; i += stride) {
    int n = i >> 8, k = i & 255;
    float v = 0.0f;
    if (n < 70) v = W_r[k * 70 + n];
    else if (n < 268) v = W_w[k * 198 + (n - 70)];
    wht[i] = (_Float16)v;
  }
  // wot[n*320+k] = W_ou[k*256+n], n<256,k<320
  for (int i = tid; i < 256 * 320; i += stride) {
    int n = i / 320, k = i % 320;
    wot[i] = (_Float16)W_ou[k * 256 + n];
  }
}

// ---------------- prep: batch-invariant scalar vectors ----------------
__global__ __launch_bounds__(256) void ntm_prep_scalar(
    const float* __restrict__ h0, const float* __restrict__ m0,
    const float* __restrict__ R0, const float* __restrict__ A0,
    const float* __restrict__ W_in, const float* __restrict__ b_in,
    const float* __restrict__ Wx, const float* __restrict__ Wh,
    const float* __restrict__ b_lstm,
    float* __restrict__ gconst, float* __restrict__ aprev, float* __restrict__ mnorm)
{
  __shared__ float hrow[256];
  __shared__ float red[256];
  int t = threadIdx.x;
  // h_row = R0 @ W_in + b_in
  float acc = b_in[t];
  for (int m = 0; m < MDIM; ++m) acc += R0[m] * W_in[m * 256 + t];
  hrow[t] = acc;
  // m_norm[n] = ||m0[n,:]||
  float ms = 0.0f;
  for (int m = 0; m < MDIM; ++m) { float v = m0[t * MDIM + m]; ms += v * v; }
  mnorm[t] = sqrtf(ms);
  __syncthreads();
  // gconst = h_row @ Wx[256:512,:] + h0 @ Wh + b_lstm
  for (int it = 0; it < 4; ++it) {
    int j = it * 256 + t;
    float g = b_lstm[j];
    for (int u = 0; u < 256; ++u) g += hrow[u] * Wx[(256 + u) * 1024 + j];
    for (int u = 0; u < 256; ++u) g += h0[u] * Wh[u * 1024 + j];
    gconst[j] = g;
  }
  // a_prev[hd] = softmax(A0[hd])
  for (int hd = 0; hd < 2; ++hd) {
    float x = A0[hd * 256 + t];
    red[t] = x; __syncthreads();
    for (int o = 128; o > 0; o >>= 1) { if (t < o) red[t] = fmaxf(red[t], red[t + o]); __syncthreads(); }
    float mx = red[0]; __syncthreads();
    float e = __expf(x - mx);
    red[t] = e; __syncthreads();
    for (int o = 128; o > 0; o >>= 1) { if (t < o) red[t] += red[t + o]; __syncthreads(); }
    float sm = red[0]; __syncthreads();
    aprev[hd * 256 + t] = e / sm;
  }
}

// ---------------- fused main kernel: 16 batch rows / block ----------------
struct SMem {
  float     m0[NSLOT * MDIM];     // 64 KB  shared memory matrix
  _Float16  Xh[16 * 256];         // 8 KB   X tile f16
  _Float16  Hh[16 * 256];         // 8 KB   h_t tile f16
  float     heads[16 * HPAD];     // 17 KB  head params f32
  float     Aw[2][16][NSLOT];     // 32 KB  addressing weights
  _Float16  Rh[16 * MDIM];        // 2 KB   read vectors f16
  float     kbuf[8][MDIM];        // 2 KB   per-wave key
  float     gc[1024];             // gate consts
  float     mn[NSLOT];            // memory row norms
  float     ap[2 * NSLOT];        // prev addressing
  float     ev[MDIM], av[MDIM];   // erase / add staging
};

__global__ __launch_bounds__(256) void ntm_main(
    const float* __restrict__ X, const float* __restrict__ c0,
    const float* __restrict__ m0g,
    const float* __restrict__ b_r, const float* __restrict__ b_w,
    const float* __restrict__ b_ou,
    const float* __restrict__ gconst, const float* __restrict__ aprev,
    const float* __restrict__ mnorm,
    const _Float16* __restrict__ WxT, const _Float16* __restrict__ WhT,
    const _Float16* __restrict__ WoT,
    float* __restrict__ y_out, float* __restrict__ h_out,
    float* __restrict__ c_out, float* __restrict__ m_out)
{
  __shared__ SMem s;
  const int t = threadIdx.x;
  const int lane = t & 31, w = t >> 5;
  const int col = lane & 15, rh = lane >> 4;
  const int b0 = blockIdx.x * 16;

  // ---- stage 0: issue async m0 -> LDS DMA (not needed until stage 3),
  //      stage X->f16 and small constants through registers.
#if USE_ASYNC_LDS
#pragma unroll
  for (int i = 0; i < 16; ++i) {
    const float4* src = ((const float4*)m0g) + i * 256 + t;
    float4* dst = ((float4*)s.m0) + i * 256 + t;
    __builtin_amdgcn_global_load_async_to_lds_b128(
        (gvp4)src, (lvp4)dst, 0, 0);
  }
#else
  for (int i = 0; i < 16; ++i)
    ((float4*)s.m0)[i * 256 + t] = ((const float4*)m0g)[i * 256 + t];
#endif
  for (int i = 0; i < 16; ++i)
    s.Xh[i * 256 + t] = (_Float16)X[(size_t)(b0 + i) * 256 + t];
  for (int i = 0; i < 4; ++i) s.gc[i * 256 + t] = gconst[i * 256 + t];
  s.mn[t] = mnorm[t];
  s.ap[t] = aprev[t];
  s.ap[256 + t] = aprev[256 + t];
  __syncthreads();   // covers Xh / gc / mn / ap (m0 waited before stage 3)

  // ---- stage 1: gates = X @ WxTop + gconst; LSTM pointwise
  // wave w owns u in [w*32, w*32+32): 4 gates x 2 col-subtiles, K=256
  v8f acc[4][2];
#pragma unroll
  for (int gi = 0; gi < 4; ++gi)
#pragma unroll
    for (int tt = 0; tt < 2; ++tt) acc[gi][tt] = (v8f){};
  const int u0 = w * 32;
  for (int kt = 0; kt < 8; ++kt) {
    v16h a = lds_a16(s.Xh, 256, kt * 32, lane);
#pragma unroll
    for (int gi = 0; gi < 4; ++gi)
#pragma unroll
      for (int tt = 0; tt < 2; ++tt) {
        v16h bfrag = glb_b16(WxT, 256, gi * 256 + u0 + tt * 16, kt * 32, lane);
        acc[gi][tt] = wmma16(a, bfrag, acc[gi][tt]);
      }
  }
#pragma unroll
  for (int tt = 0; tt < 2; ++tt) {
    int u = u0 + tt * 16 + col;
    float c0v = c0[u];
    float gci = s.gc[u], gcf = s.gc[256 + u], gcg = s.gc[512 + u], gco = s.gc[768 + u];
#pragma unroll
    for (int r = 0; r < 8; ++r) {
      int row = r + 8 * rh;
      size_t gidx = (size_t)(b0 + row) * 256 + u;
      float iv = sigf(acc[0][tt][r] + gci);
      float fv = sigf(acc[1][tt][r] + gcf);
      float gv = tanhf(acc[2][tt][r] + gcg);
      float ov = sigf(acc[3][tt][r] + gco);
      float cv = fv * c0v + iv * gv;
      float hv = ov * tanhf(cv);
      c_out[gidx] = cv;
      h_out[gidx] = hv;
      s.Hh[row * 256 + u] = (_Float16)hv;
    }
  }
  __syncthreads();

  // ---- stage 2: heads = h_t @ [W_r|W_w] + bias  (17 col tiles, K=256)
  for (int nt = w; nt < 17; nt += 8) {
    v8f hacc = (v8f){};
    for (int kt = 0; kt < 8; ++kt) {
      v16h a = lds_a16(s.Hh, 256, kt * 32, lane);
      v16h bfrag = glb_b16(WhT, 256, nt * 16, kt * 32, lane);
      hacc = wmma16(a, bfrag, hacc);
    }
    int n = nt * 16 + col;
    float bias = (n < 70) ? b_r[n] : ((n < 268) ? b_w[n - 70] : 0.0f);
#pragma unroll
    for (int r = 0; r < 8; ++r)
      s.heads[(r + 8 * rh) * HPAD + n] = hacc[r] + bias;
  }
#if USE_ASYNC_LDS
  __builtin_amdgcn_s_wait_asynccnt(0);   // m0 DMA complete (this wave)
#endif
  __syncthreads();                       // all waves' m0 slices + heads visible

  // ---- stage 3: addressing (2 heads x 16 rows = 32 pairs, 4 per wave)
  for (int pi = 0; pi < 4; ++pi) {
    int pair = w * 4 + pi;
    int hd = pair >> 4, r = pair & 15;
    const float* hrow = &s.heads[r * HPAD + hd * 70];
    // key + ||k||
    float ksq = 0.0f;
    for (int m = lane; m < MDIM; m += 32) {
      float kv = tanhf(hrow[m]);
      s.kbuf[w][m] = kv;
      ksq += kv * kv;
    }
    float knorm = sqrtf(wredsum(ksq));
    float beta = softplusf(hrow[64]);
    float gg = sigf(hrow[65]);
    float t0 = hrow[66], t1 = hrow[67], t2 = hrow[68];
    float m3 = fmaxf(t0, fmaxf(t1, t2));
    float e0 = __expf(t0 - m3), e1 = __expf(t1 - m3), e2 = __expf(t2 - m3);
    float si = 1.0f / (e0 + e1 + e2);
    float s0 = e0 * si, s1 = e1 * si, s2 = e2 * si;
    float gamma = softplusf(hrow[69]);
    // cosine sim -> softmax logits (rotated m-index: conflict-free LDS banks)
    float wv[8];
    float mx = -1e30f;
#pragma unroll
    for (int j = 0; j < 8; ++j) {
      int n = lane + 32 * j;
      float dot = 0.0f;
      for (int m = 0; m < MDIM; ++m) {
        int mm = (m + n) & 63;
        dot += s.m0[n * MDIM + mm] * s.kbuf[w][mm];
      }
      float Kv = dot / (knorm * s.mn[n] + 1e-8f);
      wv[j] = beta * Kv;
      mx = fmaxf(mx, wv[j]);
    }
    mx = wredmax(mx);
    float se = 0.0f;
#pragma unroll
    for (int j = 0; j < 8; ++j) { wv[j] = __expf(wv[j] - mx); se += wv[j]; }
    float inv = 1.0f / wredsum(se);
    // interpolate, stage w_g in Aw (same-wave LDS ops are in-order)
#pragma unroll
    for (int j = 0; j < 8; ++j) {
      int n = lane + 32 * j;
      s.Aw[hd][r][n] = gg * wv[j] * inv + (1.0f - gg) * s.ap[hd * 256 + n];
    }
    // circular 3-tap conv + sharpen + renormalize
    float cv[8];
    float ss = 0.0f;
#pragma unroll
    for (int j = 0; j < 8; ++j) {
      int n = lane + 32 * j;
      float c = s0 * s.Aw[hd][r][n] + s1 * s.Aw[hd][r][(n + 255) & 255]
              + s2 * s.Aw[hd][r][(n + 1) & 255];
      float sh = __expf(gamma * __logf(fmaxf(c, 1e-30f)));
      cv[j] = sh; ss += sh;
    }
    float inv2 = 1.0f / wredsum(ss);
#pragma unroll
    for (int j = 0; j < 8; ++j) s.Aw[hd][r][lane + 32 * j] = cv[j] * inv2;
  }
  __syncthreads();

  // ---- stage 4: reads R = A_r @ m0 -> f16 (A-operand for output GEMM)
  {
    int m = t & 63, rg = t >> 6;
    for (int r = rg; r < 16; r += 4) {
      float a2 = 0.0f;
      for (int n = 0; n < NSLOT; ++n) a2 += s.Aw[0][r][n] * s.m0[n * MDIM + m];
      s.Rh[r * MDIM + m] = (_Float16)a2;
    }
  }
  __syncthreads();

  // ---- stage 5: memory write-out (the HBM-bound part, 268 MB total)
  for (int r = 0; r < 16; ++r) {
    if (t < MDIM) {
      s.ev[t] = sigf(s.heads[r * HPAD + 140 + t]);
      s.av[t] = tanhf(s.heads[r * HPAD + 204 + t]);
    }
    __syncthreads();
    float4* dst = (float4*)(m_out + (size_t)(b0 + r) * (NSLOT * MDIM));
    int mb = (t & 15) * 4;
    float e0v = s.ev[mb], e1v = s.ev[mb + 1], e2v = s.ev[mb + 2], e3v = s.ev[mb + 3];
    float a0v = s.av[mb], a1v = s.av[mb + 1], a2v = s.av[mb + 2], a3v = s.av[mb + 3];
#pragma unroll 4
    for (int i = 0; i < 16; ++i) {
      int n = i * 16 + (t >> 4);
      float wn = s.Aw[1][r][n];
      float4 mv = *(const float4*)&s.m0[n * MDIM + mb];
      float4 o;
      o.x = mv.x * (1.0f - wn * e0v) + wn * a0v;
      o.y = mv.y * (1.0f - wn * e1v) + wn * a1v;
      o.z = mv.z * (1.0f - wn * e2v) + wn * a2v;
      o.w = mv.w * (1.0f - wn * e3v) + wn * a3v;
      dst[i * 256 + t] = o;
    }
    __syncthreads();
  }

  // ---- stage 6: y = clip([h_t | R] @ W_ou + b_ou)  (K=320: 8 Hh + 2 Rh steps)
  for (int nt = w; nt < 16; nt += 8) {
    v8f ya = (v8f){};
    for (int kt = 0; kt < 10; ++kt) {
      v16h a;
      if (kt < 8) a = lds_a16(s.Hh, 256, kt * 32, lane);
      else        a = lds_a16(s.Rh, MDIM, (kt - 8) * 32, lane);
      v16h bfrag = glb_b16(WoT, 320, nt * 16, kt * 32, lane);
      ya = wmma16(a, bfrag, ya);
    }
    int n = nt * 16 + col;
    float bias = b_ou[n];
#pragma unroll
    for (int r = 0; r < 8; ++r) {
      float yv = ya[r] + bias;
      yv = fminf(fmaxf(yv, -20.0f), 20.0f);
      y_out[(size_t)(b0 + r + 8 * rh) * 256 + n] = yv;
    }
  }
}

// ---------------- launcher ----------------
extern "C" void kernel_launch(void* const* d_in, const int* in_sizes, int n_in,
                              void* d_out, int out_size, void* d_ws, size_t ws_size,
                              hipStream_t stream) {
  const float* X      = (const float*)d_in[0];
  const float* h0     = (const float*)d_in[1];
  const float* c0     = (const float*)d_in[2];
  const float* m0     = (const float*)d_in[3];
  const float* R0     = (const float*)d_in[4];
  const float* A0     = (const float*)d_in[5];
  const float* W_in   = (const float*)d_in[6];
  const float* b_in   = (const float*)d_in[7];
  const float* Wx     = (const float*)d_in[8];
  const float* Wh     = (const float*)d_in[9];
  const float* b_lstm = (const float*)d_in[10];
  const float* W_r    = (const float*)d_in[11];
  const float* b_r    = (const float*)d_in[12];
  const float* W_w    = (const float*)d_in[13];
  const float* b_w    = (const float*)d_in[14];
  const float* W_ou   = (const float*)d_in[15];
  const float* b_ou   = (const float*)d_in[16];

  // workspace layout
  float* gconst = (float*)d_ws;            // 1024 f32
  float* aprev  = gconst + 1024;           // 512 f32
  float* mnorm  = aprev + 512;             // 256 f32
  _Float16* wxt = (_Float16*)((char*)d_ws + 8192);   // 1024*256 f16
  _Float16* wht = wxt + 1024 * 256;                  // 272*256 f16
  _Float16* wot = wht + HPAD * 256;                  // 256*320 f16

  float* y_out = (float*)d_out;
  float* h_out = y_out + (size_t)BATCH * UDIM;
  float* c_out = h_out + (size_t)BATCH * UDIM;
  float* m_out = c_out + (size_t)BATCH * UDIM;

  ntm_prep_w<<<512, 256, 0, stream>>>(Wx, W_r, W_w, W_ou, wxt, wht, wot);
  ntm_prep_scalar<<<1, 256, 0, stream>>>(h0, m0, R0, A0, W_in, b_in, Wx, Wh,
                                         b_lstm, gconst, aprev, mnorm);
  ntm_main<<<BATCH / 16, 256, 0, stream>>>(X, c0, m0, b_r, b_w, b_ou,
                                           gconst, aprev, mnorm,
                                           wxt, wht, wot,
                                           y_out, h_out, c_out, m_out);
}
